// ObjectRepresentation_22617297781280
// MI455X (gfx1250) — matrix operand order repaired and checked by reading, not compiled, not run on hardware
//
#include <hip/hip_runtime.h>
#include <hip/hip_bf16.h>
#include <math.h>

typedef _Float16 f16;
typedef __attribute__((ext_vector_type(16))) _Float16 v16h;
typedef __attribute__((ext_vector_type(8)))  _Float16 v8h;
typedef __attribute__((ext_vector_type(8)))  float    v8f;

#define HH    512
#define WWID  512
#define HWPIX (512*512)
#define CCH   12
#define NCLS  4
#define NFR   4
#define TPX   128            // pixels per workgroup (2 M-tiles per wave, 4 waves)

// class gate conv: Cin=15 padded to 16 -> 2 groups/cell, 9 cells = 18 real groups
#define CLS_RSTR   16
#define CLS_NG     18
#define KPC        160       // 20 groups * 8 = 5 k-steps
// general gate conv: Cin=48 -> 6 groups/cell, 9 cells = 54 real groups
#define GEN_RSTR   48
#define GEN_NG     54
#define KPG        448       // 56 groups * 8 = 14 k-steps

__device__ __forceinline__ float hsig(float x) {
    return fminf(fmaxf(fmaf(x, 0.2f, 0.5f), 0.0f), 1.0f);
}

__device__ __forceinline__ v16h combine16(v8h a, v8h b) {
    v16h r;
#pragma unroll
    for (int i = 0; i < 8; ++i) { r[i] = a[i]; r[i + 8] = b[i]; }
    return r;
}

// Address of the 8-f16 contiguous run for permuted-K group g, pixel-row p.
// Group g -> cell = g/GPC (kh=cell/3, kw=cell%3), sub-chunk = g%GPC.
// Halo tile layout: raw[(kh*130 + p + kw)*RSTR + sub*8]; pad groups -> zpad.
__device__ __forceinline__ const f16* cls_gaddr(const f16* raw, const f16* zpad,
                                                int g, int p) {
    if (g >= CLS_NG) return zpad;
    int cell = g >> 1, sub = g & 1;
    int kh = cell / 3, kw = cell - kh * 3;
    return raw + ((kh * 130 + p + kw) << 4) + (sub << 3);
}
__device__ __forceinline__ const f16* gen_gaddr(const f16* raw, const f16* zpad,
                                                int g, int p) {
    if (g >= GEN_NG) return zpad;
    int cell = g / 6, sub = g - cell * 6;
    int kh = cell / 3, kw = cell - kh * 3;
    return raw + (kh * 130 + p + kw) * GEN_RSTR + sub * 8;
}

// ---------------------------------------------------------------- utilities
__global__ void zero_pooled_kernel(float* __restrict__ pooled) {
    pooled[threadIdx.x] = 0.0f;   // 4 frames * 64 channels
}

// ------------------------------------------------- classifier: conv + mean
__global__ __launch_bounds__(256) void classify_conv_kernel(
    const float* __restrict__ x, const float* __restrict__ cw,
    const float* __restrict__ cb, float* __restrict__ pooled)
{
    __shared__ float Wl[27 * 64];
    __shared__ float Bl[64];
    __shared__ float sred[64];
    const int tid = threadIdx.x;
    const int f = blockIdx.y;
    for (int i = tid; i < 27 * 64; i += 256) Wl[i] = cw[i];   // (3,3,3,64) HWIO
    if (tid < 64) { Bl[tid] = cb[tid]; sred[tid] = 0.0f; }
    __syncthreads();

    const int o  = blockIdx.x * 256 + tid;     // stride-4 SAME => interior only
    const int oh = o >> 7, ow = o & 127;
    float in[27];
    int j = 0;
#pragma unroll
    for (int kh = 0; kh < 3; ++kh)
#pragma unroll
        for (int kw = 0; kw < 3; ++kw)
#pragma unroll
            for (int ci = 0; ci < 3; ++ci)
                in[j++] = x[(((size_t)(4 * oh + kh)) * WWID + (4 * ow + kw)) * CCH + 3 * f + ci];

    for (int co = 0; co < 64; ++co) {
        float s = Bl[co];
#pragma unroll
        for (int t = 0; t < 27; ++t) s = fmaf(in[t], Wl[t * 64 + co], s);
        atomicAdd(&sred[co], fmaxf(s, 0.0f));
    }
    __syncthreads();
    if (tid < 64) atomicAdd(&pooled[f * 64 + tid], sred[tid]);
}

// ------------------------------------------------- classifier: dense head
__global__ __launch_bounds__(256) void classify_head_kernel(
    const float* __restrict__ pooled, const float* __restrict__ d1w,
    const float* __restrict__ d1b, const float* __restrict__ d2w,
    const float* __restrict__ d2b, int* __restrict__ labels)
{
    __shared__ float pl[64];
    __shared__ float hid[1024];
    __shared__ float lg[4];
    const int tid = threadIdx.x;
    const int f = blockIdx.x;
    if (tid < 64) pl[tid] = pooled[f * 64 + tid] * (1.0f / 16384.0f);  // mean
    if (tid < 4)  lg[tid] = 0.0f;
    __syncthreads();
    for (int jj = tid; jj < 1024; jj += 256) {
        float s = d1b[jj];
        for (int ci = 0; ci < 64; ++ci) s = fmaf(pl[ci], d1w[ci * 1024 + jj], s);
        hid[jj] = fmaxf(s, 0.0f);
    }
    __syncthreads();
    float p0 = 0, p1 = 0, p2 = 0, p3 = 0;
    for (int jj = tid; jj < 1024; jj += 256) {
        float hv = hid[jj];
        p0 = fmaf(hv, d2w[jj * 4 + 0], p0);
        p1 = fmaf(hv, d2w[jj * 4 + 1], p1);
        p2 = fmaf(hv, d2w[jj * 4 + 2], p2);
        p3 = fmaf(hv, d2w[jj * 4 + 3], p3);
    }
    atomicAdd(&lg[0], p0); atomicAdd(&lg[1], p1);
    atomicAdd(&lg[2], p2); atomicAdd(&lg[3], p3);
    __syncthreads();
    if (tid == 0) {  // softmax is monotone: argmax(logits)
        float best = lg[0] + d2b[0]; int bi = 0;
        for (int c = 1; c < 4; ++c) {
            float v = lg[c] + d2b[c];
            if (v > best) { best = v; bi = c; }
        }
        labels[f] = bi;
    }
}

// ------------- pack gate weights into f16 B [48][Kpad], permuted-K layout
// n = gate*12 + co (i,f,o,c);  k' = group*8 + j,
// class: group<18: cell=group/2, ci=(group%2)*8+j (ci 15 is zero pad)
// gen:   group<54: cell=group/6, ci=(group%6)*8+j
__global__ __launch_bounds__(256) void pack_weights_kernel(
    const float* cwi, const float* cwf, const float* cwo, const float* cwc,
    const float* cbi, const float* cbf, const float* cbo, const float* cbc,
    const float* gwi, const float* gwf, const float* gwo, const float* gwc,
    const float* gbi, const float* gbf, const float* gbo, const float* gbc,
    f16* __restrict__ Bcls, float* __restrict__ bias_cls,
    f16* __restrict__ Bgen, float* __restrict__ bias_gen)
{
    const int tid = threadIdx.x;
    const float* cwp[4] = { cwi, cwf, cwo, cwc };
    const float* cbp[4] = { cbi, cbf, cbo, cbc };
    const float* gwp[4] = { gwi, gwf, gwo, gwc };
    const float* gbp[4] = { gbi, gbf, gbo, gbc };

    for (int idx = tid; idx < 48 * KPC; idx += 256) {
        int n = idx / KPC, k = idx % KPC, g = n / 12, co = n % 12;
        int grp = k >> 3, j = k & 7;
        f16 v = (f16)0.0f;
        if (grp < CLS_NG) {
            int cell = grp >> 1, ci = ((grp & 1) << 3) + j;
            if (ci < 15) v = (f16)cwp[g][(cell * 15 + ci) * 12 + co];  // (3,3,15,12)
        }
        Bcls[idx] = v;
    }
    for (int idx = tid; idx < 48 * KPG; idx += 256) {
        int n = idx / KPG, k = idx % KPG, g = n / 12, co = n % 12;
        int grp = k >> 3, j = k & 7;
        f16 v = (f16)0.0f;
        if (grp < GEN_NG) {
            int cell = grp / 6, ci = (grp - cell * 6) * 8 + j;        // (3,3,48,12)
            v = (f16)gwp[g][(cell * 48 + ci) * 12 + co];
        }
        Bgen[idx] = v;
    }
    if (tid < 48) {
        bias_cls[tid] = cbp[tid / 12][tid % 12];
        bias_gen[tid] = gbp[tid / 12][tid % 12];
    }
}

// ------------------------- class-LSTM gate convs via WMMA (all 4 frames)
// Tile: row h, 128 pixels; 4 waves x 2 M-tiles; N=48 (3 wmma n-tiles);
// A fragments load straight from the halo tile (permuted-K trick).
__global__ __launch_bounds__(128) void class_gates_kernel(
    const float* __restrict__ x, const float* __restrict__ gent,
    const f16* __restrict__ Bw, const float* __restrict__ bias,
    f16* __restrict__ gates)
{
    __shared__ alignas(16) f16 raw[3 * 130 * CLS_RSTR];   // 12480 B halo tile
    __shared__ alignas(16) f16 gst[TPX * 48];             // 12288 B gate stash
    __shared__ alignas(16) f16 zpad[8];
    const int tid = threadIdx.x;
    const int f  = blockIdx.z;
    const int h  = blockIdx.y;
    const int w0 = blockIdx.x * TPX;
    if (tid < 8) zpad[tid] = (f16)0.0f;

    // halo fill: 3*130 positions x 2 chunks of 8 channels (ch15 = zero pad)
    for (int u = tid; u < 390 * 2; u += 128) {
        int c8 = u & 1, pos = u >> 1;
        int cc = pos % 130, rr = pos / 130;
        int hh = h + rr - 1, wp = w0 + cc - 1;
        v8h v;
#pragma unroll
        for (int i = 0; i < 8; ++i) v[i] = (f16)0.0f;
        if (hh >= 0 && hh < HH && wp >= 0 && wp < WWID) {
            const float* gp = gent + ((size_t)hh * WWID + wp) * CCH;
            if (c8 == 0) {
                float4 a4 = *(const float4*)gp;
                float4 b4 = *(const float4*)(gp + 4);
                v[0] = (f16)a4.x; v[1] = (f16)a4.y; v[2] = (f16)a4.z; v[3] = (f16)a4.w;
                v[4] = (f16)b4.x; v[5] = (f16)b4.y; v[6] = (f16)b4.z; v[7] = (f16)b4.w;
            } else {
                float4 a4 = *(const float4*)(gp + 8);
                const float* xp = x + ((size_t)hh * WWID + wp) * CCH + 3 * f;
                v[0] = (f16)a4.x; v[1] = (f16)a4.y; v[2] = (f16)a4.z; v[3] = (f16)a4.w;
                v[4] = (f16)xp[0]; v[5] = (f16)xp[1]; v[6] = (f16)xp[2]; v[7] = (f16)0.0f;
            }
        }
        *(v8h*)&raw[(pos << 4) + (c8 << 3)] = v;
    }
    __syncthreads();

    const int wv = tid >> 5, lane = tid & 31, lrow = lane & 15;
    const bool hi = lane >= 16;
    const int offB = hi ? 8 : 0;
    const int p0 = wv * 32 + lrow;        // M-tile 0 pixel row
    const int p1 = p0 + 16;               // M-tile 1 pixel row

    v8f acc0[3], acc1[3];
#pragma unroll
    for (int nt = 0; nt < 3; ++nt) {
        float bv = bias[nt * 16 + lrow];  // bias folded into accumulator
#pragma unroll
        for (int r = 0; r < 8; ++r) { acc0[nt][r] = bv; acc1[nt][r] = bv; }
    }

    for (int ks = 0; ks < KPC / 32; ++ks) {
        int gA = ks * 4 + (hi ? 1 : 0);   // A chunks: lo {0,2}, hi {1,3}
        v16h a0 = combine16(*(const v8h*)cls_gaddr(raw, zpad, gA,     p0),
                            *(const v8h*)cls_gaddr(raw, zpad, gA + 2, p0));
        v16h a1 = combine16(*(const v8h*)cls_gaddr(raw, zpad, gA,     p1),
                            *(const v8h*)cls_gaddr(raw, zpad, gA + 2, p1));
#pragma unroll
        for (int nt = 0; nt < 3; ++nt) {
            const f16* Bp = Bw + (size_t)(nt * 16 + lrow) * KPC + ks * 32 + offB;
            v16h b = combine16(*(const v8h*)Bp, *(const v8h*)(Bp + 16));
            acc0[nt] = __builtin_amdgcn_wmma_f32_16x16x32_f16(
                false, a0, false, b, (short)0, acc0[nt], false, false);
            acc1[nt] = __builtin_amdgcn_wmma_f32_16x16x32_f16(
                false, a1, false, b, (short)0, acc1[nt], false, false);
        }
    }

    // activations -> LDS stash (gate c -> tanh, else hard-sigmoid)
#pragma unroll
    for (int nt = 0; nt < 3; ++nt) {
        int n = nt * 16 + lrow;
        bool isC = (n >= 36);
#pragma unroll
        for (int r = 0; r < 8; ++r) {
            int m = (hi ? 8 : 0) + r;
            float va = acc0[nt][r];
            float vb = acc1[nt][r];
            gst[(wv * 32 + m) * 48 + n]      = (f16)(isC ? tanhf(va) : hsig(va));
            gst[(wv * 32 + 16 + m) * 48 + n] = (f16)(isC ? tanhf(vb) : hsig(vb));
        }
    }
    __syncthreads();
    // coalesced contiguous store of the 128x48 f16 tile
    size_t gbase = ((size_t)f * HWPIX + (size_t)h * WWID + w0) * 48;
    for (int e = tid; e < TPX * 48 / 8; e += 128)
        *(v8h*)(gates + gbase + e * 8) = *(const v8h*)&gst[e * 8];
}

// ----------------- label-dependent elementwise chain + ct buffer production
__global__ __launch_bounds__(256) void class_pointwise_kernel(
    const f16* __restrict__ gates, const int* __restrict__ labels,
    const float* __restrict__ class_c, const float* __restrict__ class_tensors,
    float* __restrict__ outp, float* __restrict__ ctbuf)
{
    const int e = blockIdx.x * 256 + threadIdx.x;
    if (e >= HWPIX * CCH) return;
    const int pix = e / CCH, co = e % CCH;

    int lab[NFR];
#pragma unroll
    for (int f = 0; f < NFR; ++f) lab[f] = labels[f];

    float cur[NCLS]   = {0, 0, 0, 0};
    float lasth[NCLS] = {0, 0, 0, 0};
    int inited[NCLS]  = {0, 0, 0, 0};
    int upd[NCLS]     = {0, 0, 0, 0};

#pragma unroll
    for (int f = 0; f < NFR; ++f) {
        int l = lab[f];
        float cp = inited[l] ? cur[l]
                             : class_c[((size_t)l * HWPIX + pix) * CCH + co];
        size_t gb = ((size_t)f * HWPIX + pix) * 48;
        float ig = (float)gates[gb + co];
        float fg = (float)gates[gb + 12 + co];
        float og = (float)gates[gb + 24 + co];
        float tc = (float)gates[gb + 36 + co];
        float cn = fg * cp + ig * tc;
        float hn = og * tanhf(cn);
        outp[(size_t)pix * 48 + f * CCH + co] = hn;   // outs concat channel-wise
        cur[l] = cn; lasth[l] = hn; inited[l] = 1; upd[l] = 1;
    }
#pragma unroll
    for (int l = 0; l < NCLS; ++l) {
        ctbuf[((size_t)l * HWPIX + pix) * CCH + co] =
            upd[l] ? lasth[l]
                   : class_tensors[((size_t)l * HWPIX + pix) * CCH + co];
    }
}

// ---------------- general ConvLSTM: 3x3 48->12 x4 gates via WMMA + pointwise
// ct buffer viewed flat as (512,512,48) NHWC (reshape is a memory no-op).
__global__ __launch_bounds__(128) void general_lstm_kernel(
    const float* __restrict__ ct, const f16* __restrict__ Bw,
    const float* __restrict__ bias, const float* __restrict__ general_c,
    float* __restrict__ outp)
{
    __shared__ alignas(16) f16 raw[3 * 130 * GEN_RSTR];   // 37440 B halo tile
    __shared__ alignas(16) f16 gst[TPX * 48];             // 12288 B gate stash
    __shared__ alignas(16) f16 zpad[8];
    const int tid = threadIdx.x;
    const int h  = blockIdx.y;
    const int w0 = blockIdx.x * TPX;
    if (tid < 8) zpad[tid] = (f16)0.0f;

    // halo fill: 3*130 positions x 6 chunks of 8 channels
    for (int u = tid; u < 390 * 6; u += 128) {
        int pos = u / 6, c8 = u - pos * 6;
        int cc = pos % 130, rr = pos / 130;
        int hh = h + rr - 1, wp = w0 + cc - 1;
        v8h v;
#pragma unroll
        for (int i = 0; i < 8; ++i) v[i] = (f16)0.0f;
        if (hh >= 0 && hh < HH && wp >= 0 && wp < WWID) {
            const float* cp = ct + ((size_t)hh * WWID + wp) * 48 + c8 * 8;
            float4 a4 = *(const float4*)cp;
            float4 b4 = *(const float4*)(cp + 4);
            v[0] = (f16)a4.x; v[1] = (f16)a4.y; v[2] = (f16)a4.z; v[3] = (f16)a4.w;
            v[4] = (f16)b4.x; v[5] = (f16)b4.y; v[6] = (f16)b4.z; v[7] = (f16)b4.w;
        }
        *(v8h*)&raw[pos * GEN_RSTR + c8 * 8] = v;
    }
    __syncthreads();

    const int wv = tid >> 5, lane = tid & 31, lrow = lane & 15;
    const bool hi = lane >= 16;
    const int offB = hi ? 8 : 0;
    const int p0 = wv * 32 + lrow;
    const int p1 = p0 + 16;

    v8f acc0[3], acc1[3];
#pragma unroll
    for (int nt = 0; nt < 3; ++nt) {
        float bv = bias[nt * 16 + lrow];
#pragma unroll
        for (int r = 0; r < 8; ++r) { acc0[nt][r] = bv; acc1[nt][r] = bv; }
    }

    for (int ks = 0; ks < KPG / 32; ++ks) {
        int gA = ks * 4 + (hi ? 1 : 0);
        v16h a0 = combine16(*(const v8h*)gen_gaddr(raw, zpad, gA,     p0),
                            *(const v8h*)gen_gaddr(raw, zpad, gA + 2, p0));
        v16h a1 = combine16(*(const v8h*)gen_gaddr(raw, zpad, gA,     p1),
                            *(const v8h*)gen_gaddr(raw, zpad, gA + 2, p1));
#pragma unroll
        for (int nt = 0; nt < 3; ++nt) {
            const f16* Bp = Bw + (size_t)(nt * 16 + lrow) * KPG + ks * 32 + offB;
            v16h b = combine16(*(const v8h*)Bp, *(const v8h*)(Bp + 16));
            acc0[nt] = __builtin_amdgcn_wmma_f32_16x16x32_f16(
                false, a0, false, b, (short)0, acc0[nt], false, false);
            acc1[nt] = __builtin_amdgcn_wmma_f32_16x16x32_f16(
                false, a1, false, b, (short)0, acc1[nt], false, false);
        }
    }

    // stash activated gates, then fused pointwise for this 128-pixel tile
#pragma unroll
    for (int nt = 0; nt < 3; ++nt) {
        int n = nt * 16 + lrow;
        bool isC = (n >= 36);
#pragma unroll
        for (int r = 0; r < 8; ++r) {
            int m = (hi ? 8 : 0) + r;
            float va = acc0[nt][r];
            float vb = acc1[nt][r];
            gst[(wv * 32 + m) * 48 + n]      = (f16)(isC ? tanhf(va) : hsig(va));
            gst[(wv * 32 + 16 + m) * 48 + n] = (f16)(isC ? tanhf(vb) : hsig(vb));
        }
    }
    __syncthreads();
    for (int e = tid; e < TPX * CCH; e += 128) {
        int p = e / CCH, co = e - p * CCH;
        float ig = (float)gst[p * 48 + co];
        float fg = (float)gst[p * 48 + 12 + co];
        float og = (float)gst[p * 48 + 24 + co];
        float tc = (float)gst[p * 48 + 36 + co];
        int pix = h * WWID + w0 + p;
        float gc = general_c[(size_t)pix * CCH + co];
        float cn = fg * gc + ig * tc;
        outp[(size_t)HWPIX * 48 + (size_t)pix * CCH + co] = og * tanhf(cn);
    }
}

// ---------------------------------------------------------------- launcher
extern "C" void kernel_launch(void* const* d_in, const int* in_sizes, int n_in,
                              void* d_out, int out_size, void* d_ws, size_t ws_size,
                              hipStream_t stream)
{
    const float* x          = (const float*)d_in[0];
    const float* clf_conv_w = (const float*)d_in[1];
    const float* clf_conv_b = (const float*)d_in[2];
    const float* clf_d1_w   = (const float*)d_in[3];
    const float* clf_d1_b   = (const float*)d_in[4];
    const float* clf_d2_w   = (const float*)d_in[5];
    const float* clf_d2_b   = (const float*)d_in[6];
    const float* cls_wi = (const float*)d_in[7];
    const float* cls_bi = (const float*)d_in[8];
    const float* gen_wi = (const float*)d_in[9];
    const float* gen_bi = (const float*)d_in[10];
    const float* cls_wf = (const float*)d_in[11];
    const float* cls_bf = (const float*)d_in[12];
    const float* gen_wf = (const float*)d_in[13];
    const float* gen_bf = (const float*)d_in[14];
    const float* cls_wo = (const float*)d_in[15];
    const float* cls_bo = (const float*)d_in[16];
    const float* gen_wo = (const float*)d_in[17];
    const float* gen_bo = (const float*)d_in[18];
    const float* cls_wc = (const float*)d_in[19];
    const float* cls_bc = (const float*)d_in[20];
    const float* gen_wc = (const float*)d_in[21];
    const float* gen_bc = (const float*)d_in[22];
    const float* class_tensors  = (const float*)d_in[23];
    /* class_h d_in[24] unused (h never read inside _conv_lstm) */
    const float* class_c        = (const float*)d_in[25];
    const float* general_tensor = (const float*)d_in[26];
    /* general_h d_in[27] unused */
    const float* general_c      = (const float*)d_in[28];

    char* ws = (char*)d_ws;
    int*   labels   = (int*)(ws + 0);
    float* pooled   = (float*)(ws + 256);
    float* bias_cls = (float*)(ws + 2048);
    float* bias_gen = (float*)(ws + 2304);
    f16*   Bcls     = (f16*)(ws + 4096);      //  48*160*2  = 15 KB
    f16*   Bgen     = (f16*)(ws + 24576);     //  48*448*2  = 42 KB
    f16*   gates    = (f16*)(ws + 131072);    //  4*HW*48*2 = 96 MB
    float* ctbuf    = (float*)(ws + 131072 + (size_t)NFR * HWPIX * 48 * sizeof(f16)); // 48 MB

    float* outp = (float*)d_out;

    zero_pooled_kernel<<<1, 256, 0, stream>>>(pooled);
    classify_conv_kernel<<<dim3(64, NFR), 256, 0, stream>>>(x, clf_conv_w, clf_conv_b, pooled);
    classify_head_kernel<<<NFR, 256, 0, stream>>>(pooled, clf_d1_w, clf_d1_b,
                                                  clf_d2_w, clf_d2_b, labels);
    pack_weights_kernel<<<1, 256, 0, stream>>>(cls_wi, cls_wf, cls_wo, cls_wc,
                                               cls_bi, cls_bf, cls_bo, cls_bc,
                                               gen_wi, gen_wf, gen_wo, gen_wc,
                                               gen_bi, gen_bf, gen_bo, gen_bc,
                                               Bcls, bias_cls, Bgen, bias_gen);
    class_gates_kernel<<<dim3(WWID / TPX, HH, NFR), 128, 0, stream>>>(x, general_tensor,
                                                                      Bcls, bias_cls, gates);
    class_pointwise_kernel<<<(HWPIX * CCH) / 256, 256, 0, stream>>>(gates, labels,
                                                                    class_c, class_tensors,
                                                                    outp, ctbuf);
    general_lstm_kernel<<<dim3(WWID / TPX, HH), 128, 0, stream>>>(ctbuf, Bgen, bias_gen,
                                                                  general_c, outp);
}
